// GCN_24489903522144
// MI455X (gfx1250) — matrix operand order repaired and checked by reading, not compiled, not run on hardware
//
#include <hip/hip_runtime.h>
#include <hip/hip_bf16.h>
#include <math.h>

// ---------------------------------------------------------------------------
// NNConv GNN fused for MI455X (gfx1250, wave32, WMMA).
//   msg[e,:] = U[e,:] @ W2'  with U[e, i*32+k] = x[src[e],i]*h[e,k],
//   W2'[(i,k),o] = w2[k, i*32+o]; b2 folded in as DIN extra K rows.
//   Each wave: 2 tiles of 16 edges -> two M=16,N=32,K'=(DIN+1)*32 GEMMs
//   via V_WMMA_F32_16X16X32_F16, sharing B fragments from LDS.
// ---------------------------------------------------------------------------

typedef _Float16 v16h __attribute__((ext_vector_type(16)));
typedef _Float16 v8h  __attribute__((ext_vector_type(8)));
typedef float    v8f  __attribute__((ext_vector_type(8)));

#define HID 32          // hidden width
#define WAVES 4         // waves per block in edge kernel

// ----------------------------- zero kernel ---------------------------------
__global__ void gnn_zero_kernel(float* __restrict__ p, int n) {
    int i = blockIdx.x * blockDim.x + threadIdx.x;
    if (i < n) p[i] = 0.0f;
}

// ----------------------------- edge kernel ---------------------------------
template <int DIN>
__global__ void __launch_bounds__(128)
gnn_edge_kernel(const float* __restrict__ x,      // [N, DIN] current features
                const int*   __restrict__ ei,     // [2, E]
                const float* __restrict__ ea,     // [E, 2]
                const float* __restrict__ w1,     // [2, 32]
                const float* __restrict__ b1,     // [32]
                const float* __restrict__ w2,     // [32, DIN*32]
                const float* __restrict__ b2,     // [DIN*32]
                float*       __restrict__ agg,    // [N, 32] (pre-zeroed)
                int nE)
{
    constexpr int NCH = DIN + 1;                  // K' chunks (+1 bias chunk)
    __shared__ v16h     fragB[NCH * 2 * 32];      // [chunk][ntile][lane]
    __shared__ _Float16 h16_s[WAVES][2][16][HID]; // f16, A-friendly reads
    __shared__ float    z_s  [WAVES][2][16][DIN];
    __shared__ int      dst_s[WAVES][2][16];

    const int lane = threadIdx.x & 31;
    const int wave = threadIdx.x >> 5;
    const int lh   = lane >> 4;                   // lane half (K-range select)
    const int lm   = lane & 15;                   // row / column within 16

    // ---- stage W2' (+ b2) into LDS, pre-swizzled into B-fragment layout ----
    // B 32x16 f16: v16h element j <-> row r = 8*(l>>4) + j for j<8,
    // r = 16 + 8*(l>>4) + (j-8) for j>=8 (matches A K map).
    constexpr int totalElems = NCH * 1024;        // NCH*2*32*16
    for (int f = threadIdx.x; f < totalElems; f += blockDim.x) {
        int c  = f >> 10;                         // chunk (i index / bias)
        int rm = f & 1023;
        int t  = rm >> 9;                         // N tile (o 0..15 / 16..31)
        int l  = (rm >> 4) & 31;                  // lane
        int j  = f & 15;                          // v16h element
        int r  = (j < 8) ? (8 * (l >> 4) + j) : (16 + 8 * (l >> 4) + (j - 8)); // K row within chunk
        int o  = t * 16 + (l & 15);               // output column
        float v;
        if (c < DIN)      v = w2[r * (DIN * HID) + c * HID + o];
        else if (r < DIN) v = b2[r * HID + o];
        else              v = 0.0f;
        ((_Float16*)fragB)[(((c * 2 + t) * 32 + l) << 4) + j] = (_Float16)v;
    }

    const int npairs = nE >> 5;                   // 32 edges per wave-iter
    const int stride = gridDim.x * WAVES;

    for (int pb = blockIdx.x * WAVES; pb < npairs; pb += stride) {
        const int  myPair = pb + wave;
        const bool act    = myPair < npairs;      // wave-uniform

        __syncthreads();                          // fragB ready / LDS reuse

        if (act) {
            const int e0 = myPair << 5;
            // dst ids: lane half lh handles sub-tile lh
            dst_s[wave][lh][lm] = ei[nE + e0 + lh * 16 + lm];
            #pragma unroll
            for (int mt = 0; mt < 2; ++mt) {
                const int e  = e0 + mt * 16 + lm;
                const int sn = ei[e];
                // gather z = x[src]: lane half lh covers features lh,lh+2,...
                #pragma unroll
                for (int i = lh; i < DIN; i += 2)
                    z_s[wave][mt][lm][i] = x[sn * DIN + i];
                // edge MLP h = relu(ea@w1+b1): lane half computes 16 k's,
                // stored as f16 (contiguous, 16B-aligned vector stores)
                const float a0 = ea[e * 2 + 0];
                const float a1 = ea[e * 2 + 1];
                v8h hv0, hv1;
                #pragma unroll
                for (int kk = 0; kk < 8; ++kk) {
                    int k0 = lh * 16 + kk;
                    int k1 = lh * 16 + 8 + kk;
                    hv0[kk] = (_Float16)fmaxf(
                        fmaf(a0, w1[k0], fmaf(a1, w1[HID + k0], b1[k0])), 0.f);
                    hv1[kk] = (_Float16)fmaxf(
                        fmaf(a0, w1[k1], fmaf(a1, w1[HID + k1], b1[k1])), 0.f);
                }
                *(v8h*)&h16_s[wave][mt][lm][lh * 16]     = hv0;
                *(v8h*)&h16_s[wave][mt][lm][lh * 16 + 8] = hv1;
            }
            // prefetch next pair's edge data into caches
            int nextPair = myPair + stride;
            if (nextPair < npairs) {
                __builtin_prefetch(&ea[(size_t)nextPair * 64 + lane * 2], 0, 1);
                __builtin_prefetch(&ei[nextPair * 32 + lane], 0, 1);
            }
        }

        __syncthreads();

        if (act) {
            // A-matrix 16x32 f16 per-lane K map: j<8 -> k=8*lh+j (run 1),
            // j>=8 -> k=16+8*lh+(j-8) (run 2): two aligned b128 LDS loads.
            v16h hf[2];
            #pragma unroll
            for (int mt = 0; mt < 2; ++mt) {
                v8h lo = *(const v8h*)&h16_s[wave][mt][lm][8 * lh];
                v8h hi = *(const v8h*)&h16_s[wave][mt][lm][16 + 8 * lh];
                #pragma unroll
                for (int j = 0; j < 8; ++j) {
                    hf[mt][j]     = lo[j];
                    hf[mt][j + 8] = hi[j];
                }
            }

            v8f acc00 = {}, acc01 = {}, acc10 = {}, acc11 = {};
            #pragma unroll
            for (int c = 0; c < NCH; ++c) {
                v16h a0, a1;
                if (c < DIN) {
                    _Float16 zc0 = (_Float16)z_s[wave][0][lm][c];
                    _Float16 zc1 = (_Float16)z_s[wave][1][lm][c];
                    #pragma unroll
                    for (int j = 0; j < 16; ++j) {
                        a0[j] = hf[0][j] * zc0;
                        a1[j] = hf[1][j] * zc1;
                    }
                } else {                           // bias chunk: U = z padded
                    #pragma unroll
                    for (int j = 0; j < 16; ++j) {
                        int k = (j < 8) ? (8 * lh + j) : (8 + 8 * lh + j);
                        a0[j] = (k < DIN) ? (_Float16)z_s[wave][0][lm][k]
                                          : (_Float16)0.0f;
                        a1[j] = (k < DIN) ? (_Float16)z_s[wave][1][lm][k]
                                          : (_Float16)0.0f;
                    }
                }
                v16h b0 = fragB[(c * 2 + 0) * 32 + lane];
                v16h b1 = fragB[(c * 2 + 1) * 32 + lane];
                acc00 = __builtin_amdgcn_wmma_f32_16x16x32_f16(
                            false, a0, false, b0, (short)0, acc00, false, false);
                acc01 = __builtin_amdgcn_wmma_f32_16x16x32_f16(
                            false, a0, false, b1, (short)0, acc01, false, false);
                acc10 = __builtin_amdgcn_wmma_f32_16x16x32_f16(
                            false, a1, false, b0, (short)0, acc10, false, false);
                acc11 = __builtin_amdgcn_wmma_f32_16x16x32_f16(
                            false, a1, false, b1, (short)0, acc11, false, false);
            }

            // C/D 16x16 f32 layout: lane n = lm, VGPR v -> row M = 8*lh + v.
            #pragma unroll
            for (int v = 0; v < 8; ++v) {
                int mrow = lh * 8 + v;
                int d0 = dst_s[wave][0][mrow];
                int d1 = dst_s[wave][1][mrow];
                atomicAdd(&agg[d0 * HID + lm],      acc00[v]);
                atomicAdd(&agg[d0 * HID + 16 + lm], acc01[v]);
                atomicAdd(&agg[d1 * HID + lm],      acc10[v]);
                atomicAdd(&agg[d1 * HID + 16 + lm], acc11[v]);
            }
        }
    }
}

// --------------------------- node update kernel -----------------------------
// agg <- relu(agg + x @ root + bias)   (agg becomes the next layer's x)
__global__ void gnn_node_kernel(const float* __restrict__ xin,
                                const float* __restrict__ root,  // [din, 32]
                                const float* __restrict__ bias,  // [32]
                                float* __restrict__ agg,         // [N, 32]
                                int nN, int din)
{
    int idx = blockIdx.x * blockDim.x + threadIdx.x;
    if (idx >= nN * HID) return;
    int nd = idx >> 5;
    int o  = idx & 31;
    float s = agg[idx] + bias[o];
    for (int i = 0; i < din; ++i)
        s = fmaf(xin[nd * din + i], root[i * HID + o], s);
    agg[idx] = fmaxf(s, 0.0f);
}

// ----------------------- final FC + log_softmax -----------------------------
__global__ void gnn_fc_lsm_kernel(const float* __restrict__ x,   // [N, 32]
                                  const float* __restrict__ w,   // [32, 4]
                                  const float* __restrict__ b,   // [4]
                                  float* __restrict__ out,       // [N, 4]
                                  int nN)
{
    int nd = blockIdx.x * blockDim.x + threadIdx.x;
    if (nd >= nN) return;
    float lg[4];
    #pragma unroll
    for (int c = 0; c < 4; ++c) {
        float s = b[c];
        for (int o = 0; o < HID; ++o)
            s = fmaf(x[nd * HID + o], w[o * 4 + c], s);
        lg[c] = s;
    }
    float mx = fmaxf(fmaxf(lg[0], lg[1]), fmaxf(lg[2], lg[3]));
    float sum = 0.0f;
    #pragma unroll
    for (int c = 0; c < 4; ++c) sum += __expf(lg[c] - mx);
    float lse = mx + __logf(sum);
    #pragma unroll
    for (int c = 0; c < 4; ++c) out[nd * 4 + c] = lg[c] - lse;
}

// ---------------------------------------------------------------------------
extern "C" void kernel_launch(void* const* d_in, const int* in_sizes, int n_in,
                              void* d_out, int out_size, void* d_ws, size_t ws_size,
                              hipStream_t stream)
{
    const float* x   = (const float*)d_in[0];
    const int*   ei  = (const int*)  d_in[1];
    const float* ea  = (const float*)d_in[2];
    const float* fcw = (const float*)d_in[21];
    const float* fcb = (const float*)d_in[22];

    const int N = in_sizes[0] / 4;    // F = 4
    const int E = in_sizes[1] / 2;    // edge_index is [2, E]

    float* bufA = (float*)d_ws;
    float* bufB = bufA + (size_t)N * HID;

    const int zeroBlocks = (N * HID + 255) / 256;
    const int nodeBlocks = (N * HID + 255) / 256;
    const int edgeBlocks = 320;

    const float* xcur = x;
    float* aggbuf = bufA;
    float* other  = bufB;

    for (int L = 0; L < 3; ++L) {
        const float* w1   = (const float*)d_in[3 + L * 6 + 0];
        const float* b1   = (const float*)d_in[3 + L * 6 + 1];
        const float* w2   = (const float*)d_in[3 + L * 6 + 2];
        const float* b2   = (const float*)d_in[3 + L * 6 + 3];
        const float* root = (const float*)d_in[3 + L * 6 + 4];
        const float* bias = (const float*)d_in[3 + L * 6 + 5];
        const int    din  = (L == 0) ? 4 : HID;

        gnn_zero_kernel<<<zeroBlocks, 256, 0, stream>>>(aggbuf, N * HID);
        if (L == 0)
            gnn_edge_kernel<4><<<edgeBlocks, 128, 0, stream>>>(
                xcur, ei, ea, w1, b1, w2, b2, aggbuf, E);
        else
            gnn_edge_kernel<HID><<<edgeBlocks, 128, 0, stream>>>(
                xcur, ei, ea, w1, b1, w2, b2, aggbuf, E);
        gnn_node_kernel<<<nodeBlocks, 256, 0, stream>>>(
            xcur, root, bias, aggbuf, N, din);

        xcur = aggbuf;                 // aggbuf now holds layer output
        float* t = aggbuf; aggbuf = other; other = t;
    }

    gnn_fc_lsm_kernel<<<(N + 255) / 256, 256, 0, stream>>>(
        xcur, fcw, fcb, (float*)d_out, N);
}
